// ImplicitGenerator3d_32804960207002
// MI455X (gfx1250) — matrix operand order repaired and checked by reading, not compiled, not run on hardware
//
#include <hip/hip_runtime.h>
#include <hip/hip_bf16.h>
#include <math.h>

#define B_BATCH   2
#define IMG       64
#define R_RAYS    4096
#define S_STEPS   12
#define RPB       49152      /* R_RAYS * S_STEPS, rows per batch  */
#define M_ROWS    98304      /* B_BATCH * RPB                     */
#define HIDN      256
#define LDC       288        /* padded hcat stride (f16)          */
#define FDIM      32
#define NCH       33         /* FDIM + sigma                      */
#define FILM_BS   1792       /* 7 * 256 film batch stride         */
#define PI_D      3.14159265358979323846

typedef __attribute__((ext_vector_type(16))) _Float16 v16h;
typedef __attribute__((ext_vector_type(8)))  _Float16 v8h;
typedef __attribute__((ext_vector_type(8)))  float    v8f;
typedef __attribute__((ext_vector_type(4)))  unsigned u32x4;

union AFrag { v16h v; v8h h[2]; };

// ---------------------------------------------------------------------------
// Camera constants (constant-folded doubles)
// ---------------------------------------------------------------------------
__device__ __forceinline__ void cam_origin(float& ox, float& oy, float& oz) {
  const double phi = PI_D * 0.5, th = PI_D * 0.5;
  ox = (float)(sin(phi) * cos(th));
  oy = (float)cos(phi);
  oz = (float)(sin(phi) * sin(th));
}

__device__ __forceinline__ float zlin_coarse(int s) {
  return 0.88f + (0.24f * (float)s) / 11.f;
}

// ---------------------------------------------------------------------------
// WMMA GEMM:  out[M x N] = epilogue(A[M x K] @ Bt^T + bias)
//   A  : f16 row-major, stride lda
//   Bt : f16 N x K row-major (pre-transposed weights); the block's
//        NT*16 x K panel is staged in LDS once, then served by ds_load.
//   EPI == 0 : FiLM sin -> f16 out (stride ldo), film batch from row index
//   EPI == 1 : sigmoid  -> f32 out (stride ldof)
// Block = 256 threads = 8 waves; wave -> 16 rows; NT n-tiles of 16 cols.
// Dynamic LDS = NT*16*K*2 bytes.
// ---------------------------------------------------------------------------
template<int NT, int EPI>
__global__ __launch_bounds__(256) void gemm_wmma_k(
    const _Float16* __restrict__ A, int lda, int K,
    const _Float16* __restrict__ Bt,
    const float* __restrict__ bias,
    const float* __restrict__ filmF, const float* __restrict__ filmP,
    _Float16* __restrict__ outH, int ldo,
    float* __restrict__ outF, int ldof)
{
  extern __shared__ _Float16 Bs[];          // NT*16 x K weight panel

  const int lane  = threadIdx.x & 31;
  const int wave  = threadIdx.x >> 5;
  const int half  = lane >> 4;
  const int rr    = lane & 15;
  const int mbase = blockIdx.x * 128 + wave * 16;
  const int nbase = blockIdx.y * (NT * 16);

  // --- cooperative stage of the weight panel into LDS (one global read) ---
  {
    const u32x4* src = (const u32x4*)(Bt + (size_t)nbase * K);
    u32x4* dst = (u32x4*)Bs;
    const int chunks = NT * 16 * K / 8;     // 16-byte chunks
    for (int i = threadIdx.x; i < chunks; i += 256) dst[i] = src[i];
  }
  __syncthreads();

  v8f acc[NT];
#pragma unroll
  for (int t = 0; t < NT; ++t)
    acc[t] = (v8f){0.f, 0.f, 0.f, 0.f, 0.f, 0.f, 0.f, 0.f};

  const _Float16* arow = A + (size_t)(mbase + rr) * lda;
  for (int k0 = 0; k0 < K; k0 += 32) {
    if (k0 + 32 < K) __builtin_prefetch(arow + k0 + 32, 0, 1);
    // A fragment: 16-bit A 16x32 layout (lanes 0-15: K 0-7 & 16-23;
    // lanes 16-31: K 8-15 & 24-31) -> two contiguous 16B loads.
    AFrag a;
    a.h[0] = *(const v8h*)(arow + k0 + 8 * half);
    a.h[1] = *(const v8h*)(arow + k0 + 16 + 8 * half);
#pragma unroll
    for (int t = 0; t < NT; ++t) {
      // B fragment: same K order as A (lanes 0-15: K 0-7 & 16-23;
      // lanes 16-31: K 8-15 & 24-31), N = lane&15 -> two 16B LDS reads.
      const _Float16* brow = Bs + (size_t)(t * 16 + rr) * K + k0 + 8 * half;
      AFrag bf;
      bf.h[0] = *(const v8h*)brow;
      bf.h[1] = *(const v8h*)(brow + 16);
      v16h bv = bf.v;
      acc[t] = __builtin_amdgcn_wmma_f32_16x16x32_f16(
          false, a.v, false, bv, (short)0, acc[t], false, false);
    }
  }

  // C/D layout: element r of v8f -> M = r + 8*half, N = lane&15
#pragma unroll
  for (int t = 0; t < NT; ++t) {
    const int n = nbase + t * 16 + rr;
#pragma unroll
    for (int r = 0; r < 8; ++r) {
      const int m = mbase + half * 8 + r;
      float v = acc[t][r] + bias[n];
      if (EPI == 0) {
        const int b = m / RPB;
        v = sinf(filmF[b * FILM_BS + n] * v + filmP[b * FILM_BS + n]);
        outH[(size_t)m * ldo + n] = (_Float16)v;
      } else {
        v = 1.f / (1.f + expf(-v));
        outF[(size_t)m * ldof + n] = v;
      }
    }
  }
}

// ---------------------------------------------------------------------------
// Weight prep: f32 (K x N) -> f16 (N x K), with zero padding for color layer
// ---------------------------------------------------------------------------
__global__ void prep_hidden_k(const float* __restrict__ Ws, _Float16* __restrict__ Wt) {
  int idx = blockIdx.x * 256 + threadIdx.x;       // 5*256*256
  int l = idx >> 16, rem = idx & 65535;
  int n = rem >> 8, k = rem & 255;
  Wt[(size_t)l * 65536 + n * 256 + k] = (_Float16)Ws[(size_t)l * 65536 + k * 256 + n];
}
__global__ void prep_col_k(const float* __restrict__ Wcol, _Float16* __restrict__ Wt) {
  int idx = blockIdx.x * 256 + threadIdx.x;       // 256 * 288
  int n = idx / 288, k = idx % 288;
  Wt[(size_t)n * 288 + k] = (k < 259) ? (_Float16)Wcol[(size_t)k * 256 + n] : (_Float16)0.f;
}
__global__ void prep_rgb_k(const float* __restrict__ Wrgb, _Float16* __restrict__ Wt) {
  int idx = blockIdx.x * 256 + threadIdx.x;       // 32 * 256
  int n = idx >> 8, k = idx & 255;
  Wt[(size_t)n * 256 + k] = (_Float16)Wrgb[(size_t)k * 32 + n];
}

// ---------------------------------------------------------------------------
// Mapping networks + FiLM parameters
// ---------------------------------------------------------------------------
__global__ void maphid_k(const float* __restrict__ z_a, const float* __restrict__ z_b,
                         const float* __restrict__ cW1, const float* __restrict__ cb1,
                         const float* __restrict__ sW1, const float* __restrict__ sb1,
                         float* __restrict__ h1) {
  int which = blockIdx.x >> 1, b = blockIdx.x & 1, j = threadIdx.x;
  const float* z = which ? (z_b + b * 256) : (z_a + b * 256);
  const float* W = which ? sW1 : cW1;
  const float* bb = which ? sb1 : cb1;
  float a = bb[j];
  for (int k = 0; k < 256; ++k) a += z[k] * W[k * 256 + j];
  h1[(which * 2 + b) * 256 + j] = (a > 0.f) ? a : 0.2f * a;
}

__global__ void film_k(const float* __restrict__ h1,
                       const float* __restrict__ cW2, const float* __restrict__ cb2,
                       const float* __restrict__ sW2, const float* __restrict__ sb2,
                       const int* __restrict__ awi,
                       float* __restrict__ film_f, float* __restrict__ film_p) {
  int b = blockIdx.x / 7, l = blockIdx.x % 7, n = threadIdx.x;
  int jf = l * 256 + n, jp = 1792 + l * 256 + n;
  const float* hc = h1 + b * 256;
  const float* hs = h1 + (2 + b) * 256;
  float fc = cb2[jf], pc = cb2[jp], fs = sb2[jf], ps = sb2[jp];
  for (int k = 0; k < 256; ++k) {
    float hck = hc[k], hsk = hs[k];
    fc += hck * cW2[(size_t)k * 3584 + jf];
    pc += hck * cW2[(size_t)k * 3584 + jp];
    fs += hsk * sW2[(size_t)k * 3584 + jf];
    ps += hsk * sW2[(size_t)k * 3584 + jp];
  }
  fc = fc * 15.f + 30.f;
  fs = fs * 15.f + 30.f;
  float w = (l >= *awi) ? 0.5f : 0.f;
  film_f[(b * 7 + l) * 256 + n] = (1.f - w) * fc + w * fs;
  film_p[(b * 7 + l) * 256 + n] = (1.f - w) * pc + w * ps;
}

__global__ void ymean_k(const float* __restrict__ z_a, const float* __restrict__ z_b,
                        const float* __restrict__ yconW, const float* __restrict__ yconb,
                        const float* __restrict__ ystyW, const float* __restrict__ ystyb,
                        float* __restrict__ ym) {
  int b = blockIdx.x, n = threadIdx.x;
  float a = yconb[n], c = ystyb[n];
  for (int k = 0; k < 256; ++k) {
    a += z_a[b * 256 + k] * yconW[k * 256 + n];
    c += z_b[b * 256 + k] * ystyW[k * 256 + n];
  }
  ym[b * 256 + n] = 0.5f * (a + c);
}

__global__ void scale_k(const float* __restrict__ ym, const float* __restrict__ Wst,
                        const float* __restrict__ bst, float* __restrict__ sc) {
  int b = blockIdx.x, c = threadIdx.x;
  float a = bst[c];
  for (int k = 0; k < 256; ++k) a += ym[b * 256 + k] * Wst[k * 32 + c];
  sc[b * 32 + c] = 1.f + tanhf(a);
}

// ---------------------------------------------------------------------------
// Ray setup
// ---------------------------------------------------------------------------
__global__ void raydirs_k(float* __restrict__ dirs) {
  int r = blockIdx.x * 256 + threadIdx.x;
  int i = r >> 6, j = r & 63;
  const double t = tan(12.0 * 0.5 * PI_D / 180.0);
  double dx = (-1.0 + 2.0 * j / 63.0) * t;
  double dy = -((-1.0 + 2.0 * i / 63.0) * t);
  double dz = -1.0;
  double inv = 1.0 / sqrt(dx * dx + dy * dy + dz * dz);
  dx *= inv; dy *= inv; dz *= inv;
  const double phi = PI_D * 0.5, th = PI_D * 0.5;
  double ox = sin(phi) * cos(th), oy = cos(phi), oz = sin(phi) * sin(th);
  double fx = -ox, fy = -oy, fz = -oz;                 // unit fwd
  double lx = fz, ly = 0.0, lz = -fx;                  // cross(up, fwd)
  double ln = sqrt(lx * lx + ly * ly + lz * lz);
  lx /= ln; ly /= ln; lz /= ln;
  double ux = fy * lz - fz * ly, uy = fz * lx - fx * lz, uz = fx * ly - fy * lx;
  double un = sqrt(ux * ux + uy * uy + uz * uz);
  ux /= un; uy /= un; uz /= un;
  // Rm columns: [-left, up2, -fwd];   dir = Rm @ d_cam
  dirs[r * 3 + 0] = (float)(-lx * dx + ux * dy - fx * dz);
  dirs[r * 3 + 1] = (float)(-ly * dx + uy * dy - fy * dz);
  dirs[r * 3 + 2] = (float)(-lz * dx + uz * dy - fz * dz);
}

__global__ void zrow_coarse_k(float* __restrict__ zrow) {
  int m = blockIdx.x * 256 + threadIdx.x;
  zrow[m] = zlin_coarse(m % S_STEPS);
}

// ---------------------------------------------------------------------------
// SIREN layer 0 (K=3, scalar) with FiLM epilogue
// ---------------------------------------------------------------------------
__global__ void layer0_k(const float* __restrict__ dirs, const float* __restrict__ zrow,
                         const float* __restrict__ W0, const float* __restrict__ b0,
                         const float* __restrict__ filmF, const float* __restrict__ filmP,
                         _Float16* __restrict__ out, int ldo) {
  const int m = blockIdx.x, n = threadIdx.x;
  const int b = m / RPB;
  const int r = (m % RPB) / S_STEPS;
  float ox, oy, oz; cam_origin(ox, oy, oz);
  const float z = zrow[m];
  const float px = ox + dirs[r * 3 + 0] * z;
  const float py = oy + dirs[r * 3 + 1] * z;
  const float pz = oz + dirs[r * 3 + 2] * z;
  float a = px * W0[n] + py * W0[256 + n] + pz * W0[512 + n] + b0[n];
  out[(size_t)m * ldo + n] =
      (_Float16)sinf(filmF[b * FILM_BS + n] * a + filmP[b * FILM_BS + n]);
}

// Fill cols 256..287 of hcat (stride LDC): dirs (f16) + zero pad
__global__ void hcat_fill_k(_Float16* __restrict__ hcat, const float* __restrict__ dirs) {
  int m = blockIdx.x * 256 + threadIdx.x;
  int r = (m % RPB) / S_STEPS;
  _Float16* row = hcat + (size_t)m * LDC;
  row[256] = (_Float16)dirs[r * 3 + 0];
  row[257] = (_Float16)dirs[r * 3 + 1];
  row[258] = (_Float16)dirs[r * 3 + 2];
#pragma unroll
  for (int k = 259; k < LDC; ++k) row[k] = (_Float16)0.f;
}

// sigma head: 256 -> 1, writes channel 32 of out33
__global__ void sigma_k(const _Float16* __restrict__ h, int ldh,
                        const float* __restrict__ Wsig,
                        const float* __restrict__ bsig, float* __restrict__ out33) {
  int m = blockIdx.x * 256 + threadIdx.x;
  const _Float16* row = h + (size_t)m * ldh;
  float a = bsig[0];
  for (int k = 0; k < 256; ++k) a += (float)row[k] * Wsig[k];
  out33[(size_t)m * NCH + 32] = a;
}

// ---------------------------------------------------------------------------
// Volumetric integration of coarse pass -> per-sample weights
// ---------------------------------------------------------------------------
__global__ void coarse_weights_k(const float* __restrict__ c33, float* __restrict__ wout) {
  int q = blockIdx.x * 256 + threadIdx.x;     // ray index (B*R)
  size_t m0 = (size_t)q * S_STEPS;
  float T = 1.f;
#pragma unroll
  for (int s = 0; s < S_STEPS; ++s) {
    float delta = (s < S_STEPS - 1) ? (zlin_coarse(s + 1) - zlin_coarse(s)) : 1e10f;
    float sg = c33[(m0 + s) * NCH + 32];
    if (sg < 0.f) sg = 0.f;
    float alpha = 1.f - expf(-delta * sg);
    wout[m0 + s] = alpha * T;
    T *= (1.f - alpha + 1e-10f);
  }
}

// sample_pdf: hierarchical fine z values (12 per ray)
__global__ void sample_pdf_k(const float* __restrict__ wbuf, float* __restrict__ fineZ) {
  int q = blockIdx.x * 256 + threadIdx.x;
  size_t m0 = (size_t)q * S_STEPS;
  float w[10], zm[11], cdf[11];
  float sum = 0.f;
#pragma unroll
  for (int i = 0; i < 10; ++i) { w[i] = wbuf[m0 + 1 + i] + 1e-5f; sum += w[i]; }
#pragma unroll
  for (int i = 0; i < 11; ++i) zm[i] = 0.5f * (zlin_coarse(i) + zlin_coarse(i + 1));
  cdf[0] = 0.f;
#pragma unroll
  for (int i = 0; i < 10; ++i) cdf[i + 1] = cdf[i] + w[i] / sum;
#pragma unroll
  for (int j = 0; j < S_STEPS; ++j) {
    float u = (float)j / 11.f;
    int ind = 0;
#pragma unroll
    for (int i = 0; i < 11; ++i) if (cdf[i] <= u) ind = i + 1;  // searchsorted right
    int below = ind - 1; if (below < 0) below = 0; if (below > 10) below = 10;
    int above = ind;     if (above > 10) above = 10;
    float denom = cdf[above] - cdf[below];
    if (denom < 1e-5f) denom = 1.f;
    fineZ[m0 + j] = zm[below] + (u - cdf[below]) / denom * (zm[above] - zm[below]);
  }
}

// ---------------------------------------------------------------------------
// Merge fine+coarse (sorted by z, stable) and integrate -> pixel features
// ---------------------------------------------------------------------------
__global__ void merge_integrate_k(const float* __restrict__ f33, const float* __restrict__ c33,
                                  const float* __restrict__ fineZ, float* __restrict__ pixbuf) {
  int q = blockIdx.x * 256 + threadIdx.x;
  size_t m0 = (size_t)q * S_STEPS;
  float az[24]; int id[24];
#pragma unroll
  for (int s = 0; s < S_STEPS; ++s) {
    az[s] = fineZ[m0 + s];            id[s] = s;
    az[12 + s] = zlin_coarse(s);      id[12 + s] = 12 + s;
  }
  for (int s = 1; s < 24; ++s) {        // stable insertion sort
    float zv = az[s]; int iv = id[s]; int t = s - 1;
    while (t >= 0 && az[t] > zv) { az[t + 1] = az[t]; id[t + 1] = id[t]; --t; }
    az[t + 1] = zv; id[t + 1] = iv;
  }
  float px[FDIM];
#pragma unroll
  for (int c = 0; c < FDIM; ++c) px[c] = 0.f;
  float T = 1.f;
  for (int s = 0; s < 24; ++s) {
    float delta = (s < 23) ? (az[s + 1] - az[s]) : 1e10f;
    int i = id[s];
    const float* row = (i < 12) ? (f33 + (m0 + i) * NCH) : (c33 + (m0 + i - 12) * NCH);
    float sg = row[32];
    if (sg < 0.f) sg = 0.f;
    float alpha = 1.f - expf(-delta * sg);
    float wgt = alpha * T;
    T *= (1.f - alpha + 1e-10f);
#pragma unroll
    for (int c = 0; c < FDIM; ++c) px[c] += wgt * row[c];
  }
#pragma unroll
  for (int c = 0; c < FDIM; ++c) pixbuf[(size_t)q * FDIM + c] = px[c];
}

// ---------------------------------------------------------------------------
// Neural renderer
// ---------------------------------------------------------------------------
__global__ void conv1_k(const float* __restrict__ pixbuf, const float* __restrict__ K1,
                        const float* __restrict__ scale, float* __restrict__ out) {
  int idx = blockIdx.x * 256 + threadIdx.x;  // B*32*64*64
  int x = idx & 63, y = (idx >> 6) & 63, o = (idx >> 12) & 31, b = idx >> 17;
  float acc = 0.f;
  for (int i = 0; i < 32; ++i)
#pragma unroll
    for (int ky = 0; ky < 3; ++ky)
#pragma unroll
      for (int kx = 0; kx < 3; ++kx) {
        int yy = y + ky - 1, xx = x + kx - 1;
        if (yy >= 0 && yy < 64 && xx >= 0 && xx < 64)
          acc += pixbuf[((size_t)(b * R_RAYS + yy * 64 + xx)) * FDIM + i] *
                 K1[((o * 32 + i) * 3 + ky) * 3 + kx];
      }
  acc *= scale[b * 32 + o];
  out[idx] = (acc > 0.f) ? acc : 0.2f * acc;
}

__global__ void conv2_k(const float* __restrict__ h, const float* __restrict__ K2,
                        const float* __restrict__ bK2, float* __restrict__ out_sr) {
  int idx = blockIdx.x * 256 + threadIdx.x;  // B*8*128*128
  int x = idx & 127, y = (idx >> 7) & 127, o = (idx >> 14) & 7, b = idx >> 17;
  float acc = bK2[o];
  for (int i = 0; i < 32; ++i)
#pragma unroll
    for (int ky = 0; ky < 3; ++ky)
#pragma unroll
      for (int kx = 0; kx < 3; ++kx) {
        int yy = y + ky - 1, xx = x + kx - 1;
        if (yy >= 0 && yy < 128 && xx >= 0 && xx < 128)
          acc += h[((size_t)(b * 32 + i)) * R_RAYS + (yy >> 1) * 64 + (xx >> 1)] *
                 K2[((o * 32 + i) * 3 + ky) * 3 + kx];
      }
  float sg = 1.f / (1.f + expf(-acc));
  out_sr[idx] = sg * 2.f - 1.f;
}

__global__ void out_pix3_k(const float* __restrict__ pixbuf, float* __restrict__ out) {
  int idx = blockIdx.x * 256 + threadIdx.x;  // B*3*64*64
  int p = idx & 4095, c = (idx >> 12) % 3, b = idx / (3 * R_RAYS);
  out[idx] = pixbuf[((size_t)(b * R_RAYS + p)) * FDIM + c] * 2.f - 1.f;
}

__global__ void pose_k(float* __restrict__ out) {
  if (threadIdx.x < 4) out[threadIdx.x] = (float)(PI_D * 0.5);
}

// ---------------------------------------------------------------------------
// Host side
// ---------------------------------------------------------------------------
struct Ctx {
  // inputs
  const float *z_a, *z_b, *cW1, *cb1, *cW2, *cb2, *sW1, *sb1, *sW2, *sb2;
  const float *yconW, *yconb, *ystyW, *ystyb, *W0, *b0, *Ws, *bs, *Wcol, *bcol;
  const float *Wrgb, *brgb, *Wsig, *bsig, *K1, *K2, *bK2, *Wstyle, *bstyle;
  const int* awi;
  // workspace
  _Float16 *wtHid, *wtCol, *wtRgb, *bufA /* M x 256 */, *bufB /* M x 288 */;
  float *film_f, *film_p, *h1, *ym, *sc, *dirs, *zrowC, *fineZ, *wgt;
  float *c33, *f33, *pixbuf, *conv1out;
};

static void film_pass(hipStream_t s, const Ctx& c, const float* zrow, float* out33) {
  const size_t lds256 = 128 * 256 * sizeof(_Float16);   // 64 KB
  const size_t lds288 = 128 * 288 * sizeof(_Float16);   // 72 KB (of 320 KB WGP LDS)
  const size_t ldsRgb = 32 * 256 * sizeof(_Float16);    // 16 KB

  layer0_k<<<M_ROWS, 256, 0, s>>>(c.dirs, zrow, c.W0, c.b0, c.film_f, c.film_p,
                                  c.bufA, 256);
  // hidden layers 1..4: ping-pong at tight stride 256
  for (int l = 1; l <= 4; ++l) {
    const _Float16* in = (l & 1) ? c.bufA : c.bufB;
    _Float16* out = (l & 1) ? c.bufB : c.bufA;
    gemm_wmma_k<8, 0><<<dim3(768, 2), 256, lds256, s>>>(
        in, 256, 256, c.wtHid + (size_t)(l - 1) * 65536, c.bs + (l - 1) * 256,
        c.film_f + l * 256, c.film_p + l * 256, out, 256, nullptr, 0);
  }
  // layer 5: bufA -> bufB used as hcat (stride 288, cols 0..255)
  gemm_wmma_k<8, 0><<<dim3(768, 2), 256, lds256, s>>>(
      c.bufA, 256, 256, c.wtHid + (size_t)4 * 65536, c.bs + 4 * 256,
      c.film_f + 5 * 256, c.film_p + 5 * 256, c.bufB, LDC, nullptr, 0);
  hcat_fill_k<<<384, 256, 0, s>>>(c.bufB, c.dirs);
  sigma_k<<<384, 256, 0, s>>>(c.bufB, LDC, c.Wsig, c.bsig, out33);
  // color layer: hcat (K=288) -> bufA (stride 256)
  gemm_wmma_k<8, 0><<<dim3(768, 2), 256, lds288, s>>>(
      c.bufB, LDC, 288, c.wtCol, c.bcol,
      c.film_f + 6 * 256, c.film_p + 6 * 256, c.bufA, 256, nullptr, 0);
  // rgb head: bufA -> out33 channels 0..31 with sigmoid
  gemm_wmma_k<2, 1><<<dim3(768, 1), 256, ldsRgb, s>>>(
      c.bufA, 256, 256, c.wtRgb, c.brgb,
      nullptr, nullptr, nullptr, 0, out33, NCH);
}

extern "C" void kernel_launch(void* const* d_in, const int* in_sizes, int n_in,
                              void* d_out, int out_size, void* d_ws, size_t ws_size,
                              hipStream_t stream) {
  (void)in_sizes; (void)n_in; (void)out_size; (void)ws_size;
  Ctx c;
  c.z_a = (const float*)d_in[0];   c.z_b = (const float*)d_in[1];
  c.cW1 = (const float*)d_in[2];   c.cb1 = (const float*)d_in[3];
  c.cW2 = (const float*)d_in[4];   c.cb2 = (const float*)d_in[5];
  c.sW1 = (const float*)d_in[6];   c.sb1 = (const float*)d_in[7];
  c.sW2 = (const float*)d_in[8];   c.sb2 = (const float*)d_in[9];
  c.yconW = (const float*)d_in[10]; c.yconb = (const float*)d_in[11];
  c.ystyW = (const float*)d_in[12]; c.ystyb = (const float*)d_in[13];
  c.W0 = (const float*)d_in[14];   c.b0 = (const float*)d_in[15];
  c.Ws = (const float*)d_in[16];   c.bs = (const float*)d_in[17];
  c.Wcol = (const float*)d_in[18]; c.bcol = (const float*)d_in[19];
  c.Wrgb = (const float*)d_in[20]; c.brgb = (const float*)d_in[21];
  c.Wsig = (const float*)d_in[22]; c.bsig = (const float*)d_in[23];
  c.K1 = (const float*)d_in[24];   c.K2 = (const float*)d_in[25];
  c.bK2 = (const float*)d_in[26];  c.Wstyle = (const float*)d_in[27];
  c.bstyle = (const float*)d_in[28];
  c.awi = (const int*)d_in[29];

  // workspace carve-out
  char* W = (char*)d_ws;
  size_t off = 0;
  auto take = [&](size_t bytes) { char* p = W + off; off += (bytes + 255) & ~(size_t)255; return p; };
  c.wtHid   = (_Float16*)take((size_t)5 * 256 * 256 * 2);
  c.wtCol   = (_Float16*)take((size_t)256 * 288 * 2);
  c.wtRgb   = (_Float16*)take((size_t)32 * 256 * 2);
  c.bufA    = (_Float16*)take((size_t)M_ROWS * 256 * 2);
  c.bufB    = (_Float16*)take((size_t)M_ROWS * LDC * 2);
  c.film_f  = (float*)take((size_t)2 * 7 * 256 * 4);
  c.film_p  = (float*)take((size_t)2 * 7 * 256 * 4);
  c.h1      = (float*)take((size_t)4 * 256 * 4);
  c.ym      = (float*)take((size_t)2 * 256 * 4);
  c.sc      = (float*)take((size_t)2 * 32 * 4);
  c.dirs    = (float*)take((size_t)R_RAYS * 3 * 4);
  c.zrowC   = (float*)take((size_t)M_ROWS * 4);
  c.fineZ   = (float*)take((size_t)M_ROWS * 4);
  c.wgt     = (float*)take((size_t)M_ROWS * 4);
  c.c33     = (float*)take((size_t)M_ROWS * NCH * 4);
  c.f33     = (float*)take((size_t)M_ROWS * NCH * 4);
  c.pixbuf  = (float*)take((size_t)B_BATCH * R_RAYS * FDIM * 4);
  c.conv1out= (float*)take((size_t)B_BATCH * 32 * R_RAYS * 4);

  float* out_pix3 = (float*)d_out;                    // B*3*64*64  = 24576
  float* out_sr   = out_pix3 + 24576;                 // B*8*128*128 = 262144
  float* out_pose = out_sr + 262144;                  // B*2 = 4

  // --- setup ---
  prep_hidden_k<<<1280, 256, 0, stream>>>(c.Ws, c.wtHid);
  prep_col_k<<<288, 256, 0, stream>>>(c.Wcol, c.wtCol);
  prep_rgb_k<<<32, 256, 0, stream>>>(c.Wrgb, c.wtRgb);
  raydirs_k<<<16, 256, 0, stream>>>(c.dirs);
  zrow_coarse_k<<<384, 256, 0, stream>>>(c.zrowC);
  maphid_k<<<4, 256, 0, stream>>>(c.z_a, c.z_b, c.cW1, c.cb1, c.sW1, c.sb1, c.h1);
  film_k<<<14, 256, 0, stream>>>(c.h1, c.cW2, c.cb2, c.sW2, c.sb2, c.awi, c.film_f, c.film_p);
  ymean_k<<<2, 256, 0, stream>>>(c.z_a, c.z_b, c.yconW, c.yconb, c.ystyW, c.ystyb, c.ym);
  scale_k<<<2, 32, 0, stream>>>(c.ym, c.Wstyle, c.bstyle, c.sc);

  // --- coarse pass ---
  film_pass(stream, c, c.zrowC, c.c33);
  coarse_weights_k<<<32, 256, 0, stream>>>(c.c33, c.wgt);
  sample_pdf_k<<<32, 256, 0, stream>>>(c.wgt, c.fineZ);

  // --- fine pass ---
  film_pass(stream, c, c.fineZ, c.f33);

  // --- merge + integrate ---
  merge_integrate_k<<<32, 256, 0, stream>>>(c.f33, c.c33, c.fineZ, c.pixbuf);

  // --- outputs ---
  out_pix3_k<<<96, 256, 0, stream>>>(c.pixbuf, out_pix3);
  conv1_k<<<1024, 256, 0, stream>>>(c.pixbuf, c.K1, c.sc, c.conv1out);
  conv2_k<<<1024, 256, 0, stream>>>(c.conv1out, c.K2, c.bK2, out_sr);
  pose_k<<<1, 4, 0, stream>>>(out_pose);
}
